// CostTokenizer_17738214932873
// MI455X (gfx1250) — matrix operand order repaired and checked by reading, not compiled, not run on hardware
//
#include <hip/hip_runtime.h>

typedef _Float16 v16h __attribute__((ext_vector_type(16)));
typedef _Float16 v8h  __attribute__((ext_vector_type(8)));
typedef float    v8f  __attribute__((ext_vector_type(8)));

// One fused kernel per pyramid level.
//   f    : [5, 2, C, H, W] fp32 features (T=5, B=2)
//   wmat : [192, 49] fp32 projection weights
//   bias : [192] fp32
//   out  : [4, 2, 192, H, W] fp32 tokens
// Grid: 8 images (4 pairs x 2 batch) x (H/16)*(W/16) tiles. Block: 256 = 8 waves.
template <int C, int H, int W>
__global__ __launch_bounds__(256) void cost_tok_kernel(
    const float* __restrict__ f, const float* __restrict__ wmat,
    const float* __restrict__ bias, float* __restrict__ out) {
  static_assert((H & (H - 1)) == 0 && (W & (W - 1)) == 0, "pow2 dims");
  constexpr int TS = 16;          // pixel tile side
  constexpr int HS = TS + 6;      // halo side (radius 3 each way) = 22
  constexpr int HALO = HS * HS;   // 484 spatial halo points
  constexpr int K = 49;           // displacement channels
  constexpr int KP = 64;          // K padded for 2 x k32 WMMA steps
  constexpr int D = 192;          // token dim
  constexpr int CSTR = 72;        // corr LDS row stride (halves, 16B aligned)
  constexpr int WSTR = 72;        // weight LDS row stride (halves)

  // fb-halo staging (phase 1, 4*484 floats = 7744B) is unioned with the
  // fp16 weight tile (phase 2, 192*72*2 = 27648B): weights staged after
  // the last halo use. Total static LDS ~= 65 KB.
  __shared__ __align__(16) unsigned char s_mix[D * WSTR * 2];
  __shared__ __align__(16) _Float16 s_corr[256 * CSTR];
  __shared__ float s_bias[D];
  _Float16* s_w  = reinterpret_cast<_Float16*>(s_mix);
  float*    s_fb = reinterpret_cast<float*>(s_mix);  // [HALO][4] interleaved

  const int tid = threadIdx.x;
  const int tx = tid & 15, ty = tid >> 4;

  constexpr int tilesW = W / TS;
  constexpr int tilesH = H / TS;
  constexpr int tilesPerImg = tilesW * tilesH;
  const int img = blockIdx.x / tilesPerImg;      // pb = p*2 + b in [0,8)
  const int tin = blockIdx.x % tilesPerImg;
  const int h0 = (tin / tilesW) * TS;
  const int w0 = (tin % tilesW) * TS;

  const size_t plane = (size_t)H * W;
  const float* fa = f + (size_t)img * C * plane;       // f[p,  b]
  const float* fb = fa + (size_t)2 * C * plane;        // f[p+1,b]
  float* outp = out + (size_t)img * D * plane;

  if (tid < D) s_bias[tid] = bias[tid];

  // ---------------- Phase 1: local correlation (VALU) ----------------
  float acc[K];
#pragma unroll
  for (int k = 0; k < K; ++k) acc[k] = 0.f;

  const int hh = h0 + ty;
  const int ww = w0 + tx;

  for (int cb = 0; cb < C / 4; ++cb) {
    __syncthreads();  // previous iteration's halo reads complete
    // Stage 4 channels of the circularly wrapped fb halo, channel-interleaved
    // so each k-tap below is a single 16B LDS load feeding 4 FMAs.
    for (int e = tid; e < 4 * HALO; e += 256) {
      const int ch = e / HALO;
      const int s  = e - ch * HALO;
      const int r  = s / HS;
      const int cc = s - r * HS;
      const int hy = (h0 - 3 + r + H) & (H - 1);
      const int hx = (w0 - 3 + cc + W) & (W - 1);
      s_fb[s * 4 + ch] =
          fb[(size_t)(cb * 4 + ch) * plane + (size_t)hy * W + hx];
    }
    __syncthreads();

    const size_t fo = (size_t)(cb * 4) * plane + (size_t)hh * W + ww;
    const float fav0 = fa[fo];
    const float fav1 = fa[fo + plane];
    const float fav2 = fa[fo + 2 * plane];
    const float fav3 = fa[fo + 3 * plane];

    const int base = (ty + 3) * HS + (tx + 3);
#pragma unroll
    for (int dy = -3; dy <= 3; ++dy) {
#pragma unroll
      for (int dx = -3; dx <= 3; ++dx) {
        const int k = (dy + 3) * 7 + (dx + 3);
        const float4 fb4 =
            *reinterpret_cast<const float4*>(&s_fb[(base - dy * HS - dx) * 4]);
        acc[k] += fav0 * fb4.x + fav1 * fb4.y + fav2 * fb4.z + fav3 * fb4.w;
      }
    }
  }
  __syncthreads();  // halo buffer dead; s_mix becomes the weight tile

  // Spill this pixel's 49 scaled correlations as a zero-padded fp16 row.
  {
    const float scale = 1.0f / sqrtf((float)C);
    _Float16* cr = &s_corr[tid * CSTR];
#pragma unroll
    for (int k = 0; k < K; ++k) cr[k] = (_Float16)(acc[k] * scale);
#pragma unroll
    for (int k = K; k < KP; ++k) cr[k] = (_Float16)0.f;
  }
  // Stage projection weights fp16, K padded 49 -> 64.
  for (int i = tid; i < D * KP; i += 256) {
    const int d = i >> 6;
    const int k = i & 63;
    s_w[d * WSTR + k] = (k < K) ? (_Float16)wmat[d * K + k] : (_Float16)0.f;
  }
  __syncthreads();

  // -------- Phase 2: WMMA GEMM  out[192 x 256] = W[192 x 64] * corr --------
  const int lane = tid & 31;
  const int wv   = tid >> 5;   // wave 0..7
  const int hi   = lane >> 4;  // half-wave
  const int ln   = lane & 15;

  for (int t = wv * 24; t < wv * 24 + 24; ++t) {  // 12 mTiles x 16 nTiles
    const int mT = t >> 4;  // d-tile 0..11
    const int nT = t & 15;  // pixel-row tile 0..15

    v8f cacc = {};
#pragma unroll
    for (int ks = 0; ks < 2; ++ks) {
      // A frag (16x32 f16): row m = mT*16+ln; halves j<8 -> k=ks*32+hi*8+j,
      // halves j>=8 -> k = +16 (ISA 7.12.2 16-bit A layout).
      const _Float16* ap = &s_w[(mT * 16 + ln) * WSTR + ks * 32 + hi * 8];
      const v8h alo = *reinterpret_cast<const v8h*>(ap);
      const v8h ahi = *reinterpret_cast<const v8h*>(ap + 16);
      v16h afr;
#pragma unroll
      for (int j = 0; j < 8; ++j) { afr[j] = alo[j]; afr[j + 8] = ahi[j]; }

      // B frag (32x16 f16): col n = pixel nT*16+ln; halves j<8 -> k=ks*32+hi*8+j,
      // halves j>=8 -> k = +16 (same k order as A).
      const _Float16* bp = &s_corr[(nT * 16 + ln) * CSTR + ks * 32 + hi * 8];
      const v8h blo = *reinterpret_cast<const v8h*>(bp);
      const v8h bhi = *reinterpret_cast<const v8h*>(bp + 16);
      v16h bfr;
#pragma unroll
      for (int j = 0; j < 8; ++j) { bfr[j] = blo[j]; bfr[j + 8] = bhi[j]; }

      cacc = __builtin_amdgcn_wmma_f32_16x16x32_f16(
          /*neg_a=*/false, afr, /*neg_b=*/false, bfr,
          /*c_mod=*/(short)0, cacc, /*reuse_a=*/false, /*reuse_b=*/false);
    }

    // D frag: VGPR r holds M = mT*16 + hi*8 + r, N = ln -> pixel (nT, ln).
    const size_t pixoff = (size_t)(h0 + nT) * W + (w0 + ln);
#pragma unroll
    for (int r = 0; r < 8; ++r) {
      const int d = mT * 16 + hi * 8 + r;
      outp[(size_t)d * plane + pixoff] = cacc[r] + s_bias[d];
    }
  }
}

extern "C" void kernel_launch(void* const* d_in, const int* in_sizes, int n_in,
                              void* d_out, int out_size, void* d_ws,
                              size_t ws_size, hipStream_t stream) {
  (void)in_sizes; (void)n_in; (void)out_size; (void)d_ws; (void)ws_size;
  const float* f1 = (const float*)d_in[0];
  const float* f2 = (const float*)d_in[1];
  const float* f3 = (const float*)d_in[2];
  const float* w1 = (const float*)d_in[3];
  const float* b1 = (const float*)d_in[4];
  const float* w2 = (const float*)d_in[5];
  const float* b2 = (const float*)d_in[6];
  const float* w3 = (const float*)d_in[7];
  const float* b3 = (const float*)d_in[8];
  float* out = (float*)d_out;

  // out offsets: t1 = 4*2*192*128*128, t2 = 4*2*192*64*64, t3 = 4*2*192*32*32
  float* o1 = out;
  float* o2 = o1 + (size_t)25165824;
  float* o3 = o2 + (size_t)6291456;

  cost_tok_kernel<64, 128, 128><<<dim3(8 * 64), dim3(256), 0, stream>>>(f1, w1, b1, o1);
  cost_tok_kernel<128, 64, 64><<<dim3(8 * 16), dim3(256), 0, stream>>>(f2, w2, b2, o2);
  cost_tok_kernel<192, 32, 32><<<dim3(8 * 4), dim3(256), 0, stream>>>(f3, w3, b3, o3);
}
